// MambaBlock_17111149707729
// MI455X (gfx1250) — matrix operand order repaired and checked by reading, not compiled, not run on hardware
//
#include <hip/hip_runtime.h>
#include <hip/hip_bf16.h>

typedef __bf16 bf16_t;
typedef __attribute__((ext_vector_type(16))) __bf16 v16bf;
typedef __attribute__((ext_vector_type(8)))  __bf16 v8bf;
typedef __attribute__((ext_vector_type(8)))  float  v8f;

typedef int v4i __attribute__((vector_size(16)));
typedef __attribute__((address_space(1))) v4i as1_v4i;
typedef __attribute__((address_space(3))) v4i as3_v4i;

#define B_    2
#define L_    4096
#define DM_   1024
#define DS_   16
#define DC_   4
#define DI_   2048
#define DTR_  64
#define NROWS (B_ * L_)           // 8192
#define DBL_LD 128                // padded (DTR + 2*DS = 96 -> 128)

// block-tile geometry for the LDS-staged GEMM
#define BM_   128
#define BN_   256
#define BK_   32
#define ASTR_ 40                  // padded LDS row stride (elems): 80B, breaks bank cycles
#define BSTR_ 40

#if __has_builtin(__builtin_amdgcn_global_load_async_to_lds_b128)
#define HAVE_ASYNC_LDS 1
#else
#define HAVE_ASYNC_LDS 0
#endif

// ---------------------------------------------------------------------------
// helpers
// ---------------------------------------------------------------------------
__device__ __forceinline__ float silu_f(float v) {
    return v / (1.f + __expf(-v));
}
__device__ __forceinline__ float softplus_f(float v) {
    return (v > 20.f) ? v : log1pf(__expf(v));
}

// 16-byte global->LDS copy: async DMA when available, else sync through VGPRs
__device__ __forceinline__ void cp16_to_lds(const bf16_t* g, bf16_t* l) {
#if HAVE_ASYNC_LDS
    __builtin_amdgcn_global_load_async_to_lds_b128(
        (as1_v4i*)(unsigned long long)g,
        (as3_v4i*)(unsigned)(unsigned long long)l,
        0, 0);
#else
    *(v8bf*)l = *(const v8bf*)g;
#endif
}

__device__ __forceinline__ void async_wait_all() {
#if HAVE_ASYNC_LDS
#if __has_builtin(__builtin_amdgcn_s_wait_asynccnt)
    __builtin_amdgcn_s_wait_asynccnt(0);
#else
    asm volatile("s_wait_asynccnt 0x0" ::: "memory");
#endif
#endif
}

// ---------------------------------------------------------------------------
// fp32 -> bf16 conversions
// ---------------------------------------------------------------------------
__global__ __launch_bounds__(256) void f32_to_bf16_kernel(
        const float* __restrict__ in, bf16_t* __restrict__ out, long n) {
    long i = (long)blockIdx.x * blockDim.x + threadIdx.x;
    if (i < n) out[i] = (bf16_t)in[i];
}

// W_xp (96 x DI) -> zero padded (128 x DI) bf16
__global__ __launch_bounds__(256) void wxp_pad_bf16_kernel(
        const float* __restrict__ in, bf16_t* __restrict__ out) {
    long i = (long)blockIdx.x * blockDim.x + threadIdx.x;
    long n = (long)DBL_LD * DI_;
    if (i >= n) return;
    int r = (int)(i / DI_);
    int c = (int)(i % DI_);
    out[i] = (r < (DTR_ + 2 * DS_)) ? (bf16_t)in[(long)r * DI_ + c] : (bf16_t)0.f;
}

// dt_lo = dbl[:, :DTR] -> bf16 (NROWS x DTR)
__global__ __launch_bounds__(256) void dtlo_bf16_kernel(
        const float* __restrict__ dbl, bf16_t* __restrict__ out) {
    long i = (long)blockIdx.x * blockDim.x + threadIdx.x;
    long n = (long)NROWS * DTR_;
    if (i >= n) return;
    int r = (int)(i / DTR_);
    int c = (int)(i % DTR_);
    out[i] = (bf16_t)dbl[(long)r * DBL_LD + c];
}

// ---------------------------------------------------------------------------
// LayerNorm: one block per row (DM_=1024, 256 threads * float4)
// ---------------------------------------------------------------------------
__global__ __launch_bounds__(256) void ln_kernel(
        const float* __restrict__ x, const float* __restrict__ g,
        const float* __restrict__ beta, bf16_t* __restrict__ xn) {
    __shared__ float s_sum[256];
    __shared__ float s_sq[256];
    const int tid = threadIdx.x;
    const int row = blockIdx.x;
    const float4 xv = ((const float4*)(x + (size_t)row * DM_))[tid];
    float s = xv.x + xv.y + xv.z + xv.w;
    float q = xv.x * xv.x + xv.y * xv.y + xv.z * xv.z + xv.w * xv.w;
    s_sum[tid] = s;
    s_sq[tid] = q;
    __syncthreads();
    for (int off = 128; off > 0; off >>= 1) {
        if (tid < off) {
            s_sum[tid] += s_sum[tid + off];
            s_sq[tid]  += s_sq[tid + off];
        }
        __syncthreads();
    }
    const float mu   = s_sum[0] * (1.f / DM_);
    const float var  = s_sq[0] * (1.f / DM_) - mu * mu;
    const float rstd = rsqrtf(var + 1e-5f);
    const int c = tid * 4;
    bf16_t* o = xn + (size_t)row * DM_ + c;
    o[0] = (bf16_t)((xv.x - mu) * rstd * g[c + 0] + beta[c + 0]);
    o[1] = (bf16_t)((xv.y - mu) * rstd * g[c + 1] + beta[c + 1]);
    o[2] = (bf16_t)((xv.z - mu) * rstd * g[c + 2] + beta[c + 2]);
    o[3] = (bf16_t)((xv.w - mu) * rstd * g[c + 3] + beta[c + 3]);
}

// ---------------------------------------------------------------------------
// Fragment builders (ISA 7.12.2 layouts, wave32)
//   A 16x32 bf16 : lane -> row (lane&15); two contiguous 8-elem chunks at
//                  K = 8*(lane>=16) and K = 16 + 8*(lane>=16)
//   B 32x16 bf16 : lane -> col (lane&15); two contiguous 8-elem chunks at
//                  K = 8*(lane>=16) and K = 16 + 8*(lane>=16)
// ---------------------------------------------------------------------------
__device__ __forceinline__ v16bf frag_from_2x8(const bf16_t* p0, const bf16_t* p1) {
    v8bf lo = *(const v8bf*)p0;
    v8bf hi = *(const v8bf*)p1;
    v16bf f;
#pragma unroll
    for (int e = 0; e < 8; ++e) { f[e] = lo[e]; f[e + 8] = hi[e]; }
    return f;
}

// ---------------------------------------------------------------------------
// Direct-from-global WMMA GEMM (used for small-N / small-K GEMMs).
// C[M,N] = A[M,K] * B[N,K]^T ; one wave owns a 64x64 tile (4x4 subtiles).
// EPI: 0 = plain, 1 = softplus(acc + bias[col]), 2 = acc + resid[row,col]
// ---------------------------------------------------------------------------
template <int EPI>
__global__ __launch_bounds__(256) void gemm_wmma_bf16(
        const bf16_t* __restrict__ A, int lda,
        const bf16_t* __restrict__ Bmat, int ldb,
        float* __restrict__ C, int ldc,
        int M, int N, int K,
        const float* __restrict__ bias,
        const float* __restrict__ resid, int ldr) {
    const int lane   = threadIdx.x & 31;
    const int gwave  = (int)((blockIdx.x * (unsigned)blockDim.x + threadIdx.x) >> 5);
    const int tilesN = N >> 6;
    const int tm = gwave / tilesN;
    const int tn = gwave % tilesN;
    if (tm * 64 >= M) return;
    const int rowBase = tm * 64;
    const int colBase = tn * 64;

    const int frow = lane & 15;
    const int kA = (lane >> 4) << 3;
    const int kB = (lane >> 4) << 3;

    const bf16_t* aptr[4];
    const bf16_t* bptr[4];
#pragma unroll
    for (int i = 0; i < 4; ++i)
        aptr[i] = A + (size_t)(rowBase + 16 * i + frow) * lda + kA;
#pragma unroll
    for (int j = 0; j < 4; ++j)
        bptr[j] = Bmat + (size_t)(colBase + 16 * j + frow) * ldb + kB;

    v8f acc[4][4];
#pragma unroll
    for (int i = 0; i < 4; ++i)
#pragma unroll
        for (int j = 0; j < 4; ++j)
#pragma unroll
            for (int v = 0; v < 8; ++v) acc[i][j][v] = 0.f;

    for (int k0 = 0; k0 < K; k0 += 32) {
        v16bf afr[4], bfr[4];
#pragma unroll
        for (int i = 0; i < 4; ++i)
            afr[i] = frag_from_2x8(aptr[i] + k0, aptr[i] + k0 + 16);
#pragma unroll
        for (int j = 0; j < 4; ++j)
            bfr[j] = frag_from_2x8(bptr[j] + k0, bptr[j] + k0 + 16);
#pragma unroll
        for (int i = 0; i < 4; ++i)
#pragma unroll
            for (int j = 0; j < 4; ++j)
                acc[i][j] = __builtin_amdgcn_wmma_f32_16x16x32_bf16(
                        false, afr[i], false, bfr[j], (short)0, acc[i][j],
                        false, false);
    }

    const int rOff = (lane >> 4) << 3;
    const int ncol = lane & 15;
#pragma unroll
    for (int i = 0; i < 4; ++i) {
#pragma unroll
        for (int j = 0; j < 4; ++j) {
            const int col = colBase + 16 * j + ncol;
#pragma unroll
            for (int v = 0; v < 8; ++v) {
                const int row = rowBase + 16 * i + rOff + v;
                float val = acc[i][j][v];
                if (EPI == 1) val = softplus_f(val + bias[col]);
                if (EPI == 2) val += resid[(size_t)row * ldr + col];
                C[(size_t)row * ldc + col] = val;
            }
        }
    }
}

// ---------------------------------------------------------------------------
// LDS-staged, double-buffered WMMA GEMM for the big GEMMs.
// Block = 256 threads = 8 waves (2 x 4); block tile 128 x 256; K-step 32.
// Tiles staged via GLOBAL_LOAD_ASYNC_TO_LDS (ASYNCcnt) when available.
// Requires M % 128 == 0, N % 256 == 0, K % 32 == 0.
// EPI: 0 = plain store, 2 = acc + resid[row,col]
// ---------------------------------------------------------------------------
template <int EPI>
__global__ __launch_bounds__(256) void gemm_wmma_bf16_lds(
        const bf16_t* __restrict__ A, int lda,
        const bf16_t* __restrict__ Bmat, int ldb,
        float* __restrict__ C, int ldc,
        int M, int N, int K,
        const float* __restrict__ resid, int ldr) {
    __shared__ bf16_t Asm[2][BM_ * ASTR_];   // 2 x 10 KB
    __shared__ bf16_t Bsm[2][BN_ * BSTR_];   // 2 x 20 KB

    const int tid   = threadIdx.x;
    const int lane  = tid & 31;
    const int wid   = tid >> 5;
    const int waveM = wid >> 2;              // 0..1
    const int waveN = wid & 3;               // 0..3

    const int tilesN = N / BN_;
    const int bm = (int)blockIdx.x / tilesN;
    const int bn = (int)blockIdx.x % tilesN;
    const int rowBase = bm * BM_;
    const int colBase = bn * BN_;

    // per-thread copy assignments: A = 512 16B-chunks (2/thread),
    //                              B = 1024 16B-chunks (4/thread)
    const bf16_t* gA[2]; int lAo[2];
    const bf16_t* gB[4]; int lBo[4];
#pragma unroll
    for (int q = 0; q < 2; ++q) {
        const int c = tid + 256 * q;
        const int r = c >> 2, cc = (c & 3) * 8;
        gA[q]  = A + (size_t)(rowBase + r) * lda + cc;
        lAo[q] = r * ASTR_ + cc;
    }
#pragma unroll
    for (int q = 0; q < 4; ++q) {
        const int c = tid + 256 * q;
        const int r = c >> 2, cc = (c & 3) * 8;
        gB[q]  = Bmat + (size_t)(colBase + r) * ldb + cc;
        lBo[q] = r * BSTR_ + cc;
    }

    auto issue = [&](int buf, int k0) {
#pragma unroll
        for (int q = 0; q < 2; ++q) cp16_to_lds(gA[q] + k0, &Asm[buf][lAo[q]]);
#pragma unroll
        for (int q = 0; q < 4; ++q) cp16_to_lds(gB[q] + k0, &Bsm[buf][lBo[q]]);
    };

    v8f acc[4][4];
#pragma unroll
    for (int i = 0; i < 4; ++i)
#pragma unroll
        for (int j = 0; j < 4; ++j)
#pragma unroll
            for (int v = 0; v < 8; ++v) acc[i][j][v] = 0.f;

    const int frow = lane & 15;
    const int kA = (lane >> 4) << 3;
    const int kB = (lane >> 4) << 3;
    const int rowLoc = waveM * 64;
    const int colLoc = waveN * 64;

    issue(0, 0);

    const int Kiters = K / BK_;
    for (int it = 0; it < Kiters; ++it) {
        const int buf = it & 1;
        async_wait_all();
        __syncthreads();                 // tiles for `buf` visible to all waves
        if (it + 1 < Kiters) issue(buf ^ 1, (it + 1) * BK_);

        const bf16_t* As = &Asm[buf][0];
        const bf16_t* Bs = &Bsm[buf][0];
        v16bf afr[4], bfr[4];
#pragma unroll
        for (int i = 0; i < 4; ++i) {
            const bf16_t* p = As + (rowLoc + 16 * i + frow) * ASTR_ + kA;
            afr[i] = frag_from_2x8(p, p + 16);
        }
#pragma unroll
        for (int j = 0; j < 4; ++j) {
            const bf16_t* p = Bs + (colLoc + 16 * j + frow) * BSTR_ + kB;
            bfr[j] = frag_from_2x8(p, p + 16);
        }
#pragma unroll
        for (int i = 0; i < 4; ++i)
#pragma unroll
            for (int j = 0; j < 4; ++j)
                acc[i][j] = __builtin_amdgcn_wmma_f32_16x16x32_bf16(
                        false, afr[i], false, bfr[j], (short)0, acc[i][j],
                        false, false);
    }

    const int rOff = (lane >> 4) << 3;
    const int ncol = lane & 15;
#pragma unroll
    for (int i = 0; i < 4; ++i) {
#pragma unroll
        for (int j = 0; j < 4; ++j) {
            const int col = colBase + colLoc + 16 * j + ncol;
#pragma unroll
            for (int v = 0; v < 8; ++v) {
                const int row = rowBase + rowLoc + 16 * i + rOff + v;
                float val = acc[i][j][v];
                if (EPI == 2) val += resid[(size_t)row * ldr + col];
                C[(size_t)row * ldc + col] = val;
            }
        }
    }
}

// ---------------------------------------------------------------------------
// causal depthwise conv (DC=4) + silu. xz is (NROWS, 2*DI); xs = first DI cols
// ---------------------------------------------------------------------------
__global__ __launch_bounds__(256) void conv_silu_kernel(
        const float* __restrict__ xz, const float* __restrict__ w,
        const float* __restrict__ cb, float* __restrict__ xc,
        bf16_t* __restrict__ xcbf) {
    const size_t idx = (size_t)blockIdx.x * blockDim.x + threadIdx.x;
    if (idx >= (size_t)NROWS * DI_) return;
    const int d  = (int)(idx % DI_);
    const int bl = (int)(idx / DI_);
    const int b  = bl / L_;
    const int l  = bl % L_;
    float acc = cb[d];
#pragma unroll
    for (int j = 0; j < DC_; ++j) {
        const int li = l + j - (DC_ - 1);
        if (li >= 0)
            acc += xz[(size_t)(b * L_ + li) * (2 * DI_) + d] * w[d * DC_ + j];
    }
    const float v = silu_f(acc);
    xc[idx]   = v;
    xcbf[idx] = (bf16_t)v;
}

// ---------------------------------------------------------------------------
// selective scan: one thread per (b, d) channel; 16-float state in VGPRs.
// Fuses y = (scan + xc*D) * silu(z) and bf16 output for the final GEMM.
// ---------------------------------------------------------------------------
__global__ __launch_bounds__(256) void scan_kernel(
        const float* __restrict__ dt, const float* __restrict__ xc,
        const float* __restrict__ dbl, const float* __restrict__ A_log,
        const float* __restrict__ Dp, const float* __restrict__ xz,
        bf16_t* __restrict__ ybf) {
    const int t = blockIdx.x * blockDim.x + threadIdx.x;
    if (t >= B_ * DI_) return;
    const int b = t / DI_;
    const int d = t % DI_;
    float a[DS_], h[DS_];
#pragma unroll
    for (int n = 0; n < DS_; ++n) {
        a[n] = -__expf(A_log[d * DS_ + n]);
        h[n] = 0.f;
    }
    const float Dd = Dp[d];
    for (int l = 0; l < L_; ++l) {
        const size_t bl  = (size_t)b * L_ + l;
        const float dtv  = dt[bl * DI_ + d];
        const float xv   = xc[bl * DI_ + d];
        const float dtx  = dtv * xv;
        const float4* q4 = (const float4*)(dbl + bl * DBL_LD + DTR_);
        float4 Bv[4] = {q4[0], q4[1], q4[2], q4[3]};   // B_t (16)
        float4 Cv[4] = {q4[4], q4[5], q4[6], q4[7]};   // C_t (16)
        const float* Bf = (const float*)Bv;
        const float* Cf = (const float*)Cv;
        float y = 0.f;
#pragma unroll
        for (int n = 0; n < DS_; ++n) {
            h[n] = __expf(dtv * a[n]) * h[n] + dtx * Bf[n];
            y += h[n] * Cf[n];
        }
        const float zv = xz[bl * (2 * DI_) + DI_ + d];
        const float yo = (y + xv * Dd) * silu_f(zv);
        ybf[bl * DI_ + d] = (bf16_t)yo;
    }
}

// ---------------------------------------------------------------------------
// launch
// ---------------------------------------------------------------------------
extern "C" void kernel_launch(void* const* d_in, const int* in_sizes, int n_in,
                              void* d_out, int out_size, void* d_ws, size_t ws_size,
                              hipStream_t stream) {
    const float* x      = (const float*)d_in[0];
    const float* ln_g   = (const float*)d_in[1];
    const float* ln_b   = (const float*)d_in[2];
    const float* W_in   = (const float*)d_in[3];
    const float* conv_w = (const float*)d_in[4];
    const float* conv_b = (const float*)d_in[5];
    const float* W_xp   = (const float*)d_in[6];
    const float* W_dt   = (const float*)d_in[7];
    const float* b_dt   = (const float*)d_in[8];
    const float* A_log  = (const float*)d_in[9];
    const float* Dv     = (const float*)d_in[10];
    const float* W_out  = (const float*)d_in[11];
    float* out = (float*)d_out;

    char* ws = (char*)d_ws;
    size_t off = 0;
    auto alloc = [&](size_t bytes) -> char* {
        char* p = ws + off;
        off = (off + bytes + 255) & ~(size_t)255;
        return p;
    };

    bf16_t* xn_bf   = (bf16_t*)alloc((size_t)NROWS * DM_ * 2);
    bf16_t* Win_bf  = (bf16_t*)alloc((size_t)2 * DI_ * DM_ * 2);
    float*  xzf     = (float*) alloc((size_t)NROWS * 2 * DI_ * 4);
    float*  xcf     = (float*) alloc((size_t)NROWS * DI_ * 4);
    bf16_t* xc_bf   = (bf16_t*)alloc((size_t)NROWS * DI_ * 2);
    bf16_t* Wxp_bf  = (bf16_t*)alloc((size_t)DBL_LD * DI_ * 2);
    float*  dblf    = (float*) alloc((size_t)NROWS * DBL_LD * 4);
    bf16_t* dtlo_bf = (bf16_t*)alloc((size_t)NROWS * DTR_ * 2);
    bf16_t* Wdt_bf  = (bf16_t*)alloc((size_t)DI_ * DTR_ * 2);
    float*  dtf     = (float*) alloc((size_t)NROWS * DI_ * 4);
    bf16_t* y_bf    = (bf16_t*)alloc((size_t)NROWS * DI_ * 2);
    bf16_t* Wout_bf = (bf16_t*)alloc((size_t)DM_ * DI_ * 2);
    (void)ws_size; (void)in_sizes; (void)n_in; (void)out_size;

    const int TPB = 256;
    auto blocks = [&](long n) { return (int)((n + TPB - 1) / TPB); };

    // weight conversions
    {
        long n = (long)2 * DI_ * DM_;
        f32_to_bf16_kernel<<<blocks(n), TPB, 0, stream>>>(W_in, Win_bf, n);
    }
    {
        long n = (long)DI_ * DTR_;
        f32_to_bf16_kernel<<<blocks(n), TPB, 0, stream>>>(W_dt, Wdt_bf, n);
    }
    {
        long n = (long)DM_ * DI_;
        f32_to_bf16_kernel<<<blocks(n), TPB, 0, stream>>>(W_out, Wout_bf, n);
    }
    wxp_pad_bf16_kernel<<<blocks((long)DBL_LD * DI_), TPB, 0, stream>>>(W_xp, Wxp_bf);

    // 1. LayerNorm -> xn (bf16)
    ln_kernel<<<NROWS, TPB, 0, stream>>>(x, ln_g, ln_b, xn_bf);

    // 2. GEMM1 (LDS/async): xz = xn @ W_in^T   (M=8192, N=4096, K=1024)
    {
        int nblk = (NROWS / BM_) * ((2 * DI_) / BN_);
        gemm_wmma_bf16_lds<0><<<nblk, TPB, 0, stream>>>(
            xn_bf, DM_, Win_bf, DM_, xzf, 2 * DI_,
            NROWS, 2 * DI_, DM_, nullptr, 0);
    }

    // 3. depthwise conv + silu -> xc (f32 + bf16)
    conv_silu_kernel<<<blocks((long)NROWS * DI_), TPB, 0, stream>>>(
        xzf, conv_w, conv_b, xcf, xc_bf);

    // 4. GEMM2 (direct): dbl = xc @ W_xp^T  (M=8192, N=128(pad), K=2048)
    {
        int waves = (NROWS / 64) * (DBL_LD / 64);
        gemm_wmma_bf16<0><<<waves * 32 / TPB, TPB, 0, stream>>>(
            xc_bf, DI_, Wxp_bf, DI_, dblf, DBL_LD,
            NROWS, DBL_LD, DI_, nullptr, nullptr, 0);
    }

    // 5. dt_lo -> bf16
    dtlo_bf16_kernel<<<blocks((long)NROWS * DTR_), TPB, 0, stream>>>(dblf, dtlo_bf);

    // 6. GEMM3 (direct): dt = softplus(dt_lo @ W_dt^T + b_dt)  (M=8192, N=2048, K=64)
    {
        int waves = (NROWS / 64) * (DI_ / 64);
        gemm_wmma_bf16<1><<<waves * 32 / TPB, TPB, 0, stream>>>(
            dtlo_bf, DTR_, Wdt_bf, DTR_, dtf, DI_,
            NROWS, DI_, DTR_, b_dt, nullptr, 0);
    }

    // 7. selective scan (fused D-skip + silu(z) gate) -> y (bf16)
    scan_kernel<<<(B_ * DI_) / TPB, TPB, 0, stream>>>(
        dtf, xcf, dblf, A_log, Dv, xzf, y_bf);

    // 8. GEMM4 (LDS/async): out = x + y @ W_out^T  (M=8192, N=1024, K=2048)
    {
        int nblk = (NROWS / BM_) * (DM_ / BN_);
        gemm_wmma_bf16_lds<2><<<nblk, TPB, 0, stream>>>(
            y_bf, DI_, Wout_bf, DI_, out, DM_,
            NROWS, DM_, DI_, x, DM_);
    }
}
